// SynthesisLayer_35665408426097
// MI455X (gfx1250) — matrix operand order repaired and checked by reading, not compiled, not run on hardware
//
#include <hip/hip_runtime.h>

typedef __attribute__((ext_vector_type(16))) __bf16 bf16x16;
typedef __attribute__((ext_vector_type(8)))  float  f32x8;
typedef __attribute__((ext_vector_type(4)))  unsigned int u32x4;
typedef __attribute__((ext_vector_type(8)))  int i32x8;
typedef __attribute__((ext_vector_type(4)))  int i32x4;

#define CIN   128
#define COUT  128
#define INH   256
#define OH    258            // conv output spatial (pad=2)
#define KTOT  1152           // 9 taps * 128 cin, packed tap-major: k = tap*128 + cin
#define WDIM  512
#define TO    32             // final-output tile
#define DR    74             // 522-domain window per tile
#define TR    44             // 258-domain rows/cols per tile

__device__ __forceinline__ unsigned short f2bf(float f) {
    unsigned int u = __float_as_uint(f);
    u += 0x7FFFu + ((u >> 16) & 1u);            // round-to-nearest-even
    return (unsigned short)(u >> 16);
}

// ---------------------------------------------------------------- styles ----
__global__ void styles_kernel(const float* __restrict__ w,
                              const float* __restrict__ aw,
                              const float* __restrict__ ab,
                              float* __restrict__ s_norm) {
    __shared__ float red[512];
    int tid = threadIdx.x;
    int b = tid >> 7, cin = tid & 127;
    const float* wr = w + b * WDIM;
    const float* ar = aw + cin * WDIM;
    float acc = 0.f;
    for (int j = 0; j < WDIM; ++j) acc += wr[j] * ar[j];
    float st = acc * 0.04419417382415922f + ab[cin];   // 1/sqrt(512)
    red[tid] = st * st;
    __syncthreads();
    for (int off = 256; off > 0; off >>= 1) {
        if (tid < off) red[tid] += red[tid + off];
        __syncthreads();
    }
    float mean = red[0] * (1.0f / 512.0f);
    s_norm[tid] = st * rsqrtf(mean);
}

// ------------------------------------------------- modulated bf16 weights ---
// Output layout: wmod[b][cout][k], k = tap*128 + cin  (tap-major K packing)
__global__ void modw_kernel(const float* __restrict__ cw,
                            const float* __restrict__ s_norm,
                            unsigned short* __restrict__ wmod) {
    __shared__ float red[128];
    int cin  = threadIdx.x;
    int cout = blockIdx.x, b = blockIdx.y;
    const float* wr = cw + (cout * CIN + cin) * 9;
    float wl[9]; float ss = 0.f;
    for (int j = 0; j < 9; ++j) { wl[j] = wr[j]; ss += wl[j] * wl[j]; }
    red[cin] = ss; __syncthreads();
    for (int off = 64; off > 0; off >>= 1) {
        if (cin < off) red[cin] += red[cin + off];
        __syncthreads();
    }
    float wg = rsqrtf(red[0] * (1.0f / (float)KTOT));  // weight pre-norm
    __syncthreads();
    float sv = s_norm[b * CIN + cin] * wg;
    float ds = 0.f;
    for (int j = 0; j < 9; ++j) { wl[j] *= sv; ds += wl[j] * wl[j]; }
    red[cin] = ds; __syncthreads();
    for (int off = 64; off > 0; off >>= 1) {
        if (cin < off) red[cin] += red[cin + off];
        __syncthreads();
    }
    float dc = rsqrtf(red[0] + 1e-8f);                 // demod coefficient
    unsigned short* o = wmod + (size_t)(b * COUT + cout) * KTOT + cin;
    for (int j = 0; j < 9; ++j) o[j * 128] = f2bf(wl[j] * dc);
}

// -------------------------------------------- implicit-GEMM conv via WMMA ---
// Block: 256 thr = 8 waves; each wave owns 16 couts x 32 pixels (2 N-tiles,
// 2 WMMAs per K-step sharing the A fragment). K packed tap-major so the tap
// (and input row iy) is uniform per 32-wide K-step: no div/mod in im2col.
__global__ void __launch_bounds__(256) conv_wmma_kernel(
        const float* __restrict__ x,
        const unsigned short* __restrict__ wmod,
        const float* __restrict__ cbias,
        float* __restrict__ convout) {
    __shared__ unsigned short sB[32 * 32];   // [n(pixel 0..31)][k_local] bf16
    int tid  = threadIdx.x;
    int lane = tid & 31, wave = tid >> 5;
    int ox0  = blockIdx.x * 32;
    int oy   = blockIdx.y;
    int b    = blockIdx.z;
    int cout0 = wave * 16;
    int m  = lane & 15;
    int hi = lane >> 4;          // lane half: selects K sub-groups per ISA layout

    const unsigned short* arow = wmod + (size_t)(b * COUT + cout0 + m) * KTOT;
    const float* xb = x + (size_t)b * CIN * INH * INH;

    f32x8 acc0 = {0.f,0.f,0.f,0.f,0.f,0.f,0.f,0.f};
    f32x8 acc1 = {0.f,0.f,0.f,0.f,0.f,0.f,0.f,0.f};

    for (int k0 = 0; k0 < KTOT; k0 += 32) {
        int tap  = k0 >> 7;                 // uniform: 4 K-steps per tap
        int cin0 = k0 & 127;
        int kh = tap / 3, kw = tap - kh * 3;   // scalar, once per step
        int iy = oy - 2 + kh;                  // uniform row
        bool rowok = (unsigned)iy < (unsigned)INH;
        // stage im2col B tile: 32 pixels x 32 k (cin) bf16, [n][k] layout
        for (int i = tid; i < 1024; i += 256) {
            int n  = i >> 5, kl = i & 31;
            int ix = ox0 + n - 2 + kw;
            float v = 0.f;
            if (rowok && (unsigned)ix < (unsigned)INH)
                v = xb[((cin0 + kl) * INH + iy) * INH + ix];
            sB[n * 32 + kl] = f2bf(v);
        }
        __syncthreads();

        union { uint4 q[2]; bf16x16 v; } A, B0, B1;
        // A 16x32 bf16: lane<16 owns K {0..7,16..23}; lane>=16 owns {8..15,24..31}
        A.q[0] = *(const uint4*)(arow + k0 + hi * 8);
        A.q[1] = *(const uint4*)(arow + k0 + 16 + hi * 8);
        // B 32x16 bf16: lane<16 -> n=lane, K {0..7,16..23} ; lane>=16 -> n=lane-16, K {8..15,24..31}
        {
            int nn = lane & 15;
            int g  = hi * 8;
            const uint4* bp0 = (const uint4*)(&sB[nn * 32 + g]);
            const uint4* bp1 = (const uint4*)(&sB[(16 + nn) * 32 + g]);
            B0.q[0] = bp0[0]; B0.q[1] = bp0[2];
            B1.q[0] = bp1[0]; B1.q[1] = bp1[2];
        }
        acc0 = __builtin_amdgcn_wmma_f32_16x16x32_bf16(
                   false, A.v, false, B0.v, (short)0, acc0, false, false);
        acc1 = __builtin_amdgcn_wmma_f32_16x16x32_bf16(
                   false, A.v, false, B1.v, (short)0, acc1, false, false);
        __builtin_prefetch(arow + k0 + 32, 0, 3);  // next K-slab of weights
        __syncthreads();
    }

    int nn = lane & 15;
    for (int t = 0; t < 2; ++t) {
        int ox = ox0 + t * 16 + nn;
        if (ox < OH) {
            const f32x8& a = t ? acc1 : acc0;
            for (int r = 0; r < 8; ++r) {
                int mm = cout0 + r + hi * 8;   // D layout: lanes16-31 hold M+8
                convout[((size_t)(b * COUT + mm) * OH + oy) * OH + ox] = a[r] + cbias[mm];
            }
        }
    }
}

// ------------------------- fused up-FIR + lrelu/clamp + down-FIR (tiled) ----
// Interior tiles pull the 44x44 conv tile into LDS with the Tensor Data Mover
// (2D D#: 4-byte elems, tile 44x44, tensor stride 258); edge tiles (negative
// or clipped origins) use the guarded VALU path since TDM cannot zero-fill
// ahead of the tensor start.
__global__ void __launch_bounds__(256) flrelu_kernel(
        const float* __restrict__ convout,
        const float* __restrict__ upf,
        const float* __restrict__ dnf,
        float* __restrict__ out) {
    __shared__ float cbuf[TR * TR];   // conv tile   [ly][lx]   (258-domain)
    __shared__ float tbuf[TR * DR];   // after up-W  [ly][lv]
    __shared__ float ubuf[DR * DR];   // after up-H + lrelu [lu][lv] (522-domain)
    __shared__ float dbuf[DR * TO];   // after down-W [lu][lx]
    __shared__ float fu[12], fd[12];

    int tid   = threadIdx.x;
    int plane = blockIdx.y;                   // b*128 + c
    int tile  = blockIdx.x;                   // 0..63
    int ty = tile >> 3, tx = tile & 7;
    int oy0 = ty * TO, ox0 = tx * TO;
    int d0y = 2 * oy0, d0x = 2 * ox0;
    int y0 = (d0y - 9) >> 1;                  // first 258-domain row needed (floor)
    int x0 = (d0x - 9) >> 1;

    if (tid < 12) {
        fu[tid] = upf[11 - tid] * 2.0f;       // reversed, per-pass gain sqrt(4)=2
        fd[tid] = dnf[11 - tid];              // reversed, gain 1
    }
    const float* cb = convout + (size_t)plane * (OH * OH);

    bool interior = (y0 >= 0) && (x0 >= 0) && (y0 + TR <= OH) && (x0 + TR <= OH);
    if (interior) {
        if (tid < 32) {
            // Build Tensor DMA descriptor (D#), groups 0 and 1; 2D tile.
            unsigned long long ga = (unsigned long long)(const void*)(cb + y0 * OH + x0);
            unsigned int lds = (unsigned int)(size_t)(void*)&cbuf[0];  // LDS byte addr
            u32x4 g0;
            g0[0] = 1u;                                        // count=1 valid user D#
            g0[1] = lds;                                       // lds_addr
            g0[2] = (unsigned int)(ga & 0xFFFFFFFFu);          // global_addr[31:0]
            g0[3] = (unsigned int)((ga >> 32) & 0x1FFFFFFu) | (2u << 30); // addr[56:32]|type=2
            i32x8 g1;
            g1[0] = (int)(2u << 16);                           // data_size=4B, no mcast
            g1[1] = (int)((unsigned)OH << 16);                 // tensor_dim0[15:0]
            g1[2] = (int)((unsigned)OH << 16);                 // dim0[31:16]=0 | dim1[15:0]
            g1[3] = (int)((unsigned)TR << 16);                 // dim1[31:16]=0 | tile_dim0=44
            g1[4] = (int)TR;                                   // tile_dim1=44, tile_dim2=0
            g1[5] = (int)OH;                                   // tensor_dim0_stride=258
            g1[6] = 0;
            g1[7] = 0;
            i32x4 z4 = {0, 0, 0, 0};
#if __clang_major__ >= 23
            i32x8 z8 = {0, 0, 0, 0, 0, 0, 0, 0};
            __builtin_amdgcn_tensor_load_to_lds(g0, g1, z4, z4, z8, 0);
#else
            __builtin_amdgcn_tensor_load_to_lds(g0, g1, z4, z4, 0);
#endif
            __builtin_amdgcn_s_wait_tensorcnt(0);
        }
    } else {
        for (int i = tid; i < TR * TR; i += 256) {
            int ly = i / TR, lx = i - ly * TR;
            int y = y0 + ly, xx = x0 + lx;
            float v = 0.f;
            if ((unsigned)y < (unsigned)OH && (unsigned)xx < (unsigned)OH)
                v = cb[y * OH + xx];
            cbuf[i] = v;
        }
    }
    __syncthreads();

    // up-pass along W: padded/zero-inserted conv row convolved with fu
    for (int i = tid; i < TR * DR; i += 256) {
        int ly = i / DR, lv = i - ly * DR;
        int v = d0x + lv;
        float s = 0.f;
        for (int j = 0; j < 12; ++j) {
            int p = v + j;
            if (p >= 9 && !((p - 9) & 1)) {
                int xx = (p - 9) >> 1;        // OOB xx covered by zero fill / interior
                s += fu[j] * cbuf[ly * TR + (xx - x0)];
            }
        }
        tbuf[i] = s;
    }
    __syncthreads();

    // up-pass along H, then lrelu * sqrt(2), clamp +-256
    for (int i = tid; i < DR * DR; i += 256) {
        int lu = i / DR, lv = i - lu * DR;
        int u = d0y + lu;
        float s = 0.f;
        for (int j = 0; j < 12; ++j) {
            int p = u + j;
            if (p >= 9 && !((p - 9) & 1)) {
                int y = (p - 9) >> 1;
                s += fu[j] * tbuf[(y - y0) * DR + lv];
            }
        }
        s = (s >= 0.f ? s : 0.2f * s) * 1.4142135623730951f;
        s = fminf(fmaxf(s, -256.f), 256.f);
        ubuf[i] = s;
    }
    __syncthreads();

    // down-pass along W (stride 2)
    for (int i = tid; i < DR * TO; i += 256) {
        int lu = i / TO, lx = i - lu * TO;
        float s = 0.f;
        for (int j = 0; j < 12; ++j) s += fd[j] * ubuf[lu * DR + 2 * lx + j];
        dbuf[i] = s;
    }
    __syncthreads();

    // down-pass along H (stride 2) + store
    for (int i = tid; i < TO * TO; i += 256) {
        int ly = i / TO, lx = i - ly * TO;
        float s = 0.f;
        for (int j = 0; j < 12; ++j) s += fd[j] * dbuf[(2 * ly + j) * TO + lx];
        out[((size_t)plane * 256 + oy0 + ly) * 256 + ox0 + lx] = s;
    }
}

// ---------------------------------------------------------------------------
extern "C" void kernel_launch(void* const* d_in, const int* in_sizes, int n_in,
                              void* d_out, int out_size, void* d_ws, size_t ws_size,
                              hipStream_t stream) {
    const float* x   = (const float*)d_in[0];
    const float* w   = (const float*)d_in[1];
    const float* aw  = (const float*)d_in[2];
    const float* ab  = (const float*)d_in[3];
    const float* cw  = (const float*)d_in[4];
    const float* cb  = (const float*)d_in[5];
    const float* upf = (const float*)d_in[6];
    const float* dnf = (const float*)d_in[7];
    float* out = (float*)d_out;

    char* ws = (char*)d_ws;
    float* s_norm         = (float*)ws;                                   // 2 KB
    unsigned short* wmod  = (unsigned short*)(ws + 4096);                 // 1.18 MB bf16
    float* convout        = (float*)(ws + 4096 + (size_t)4 * COUT * KTOT * 2); // 136 MB

    styles_kernel<<<1, 512, 0, stream>>>(w, aw, ab, s_norm);
    modw_kernel<<<dim3(COUT, 4), CIN, 0, stream>>>(cw, s_norm, wmod);
    conv_wmma_kernel<<<dim3((OH + 31) / 32, OH, 4), 256, 0, stream>>>(x, wmod, cb, convout);
    flrelu_kernel<<<dim3(64, 512), 256, 0, stream>>>(convout, upf, dnf, out);
}
